// Block_19602230739036
// MI455X (gfx1250) — matrix operand order repaired and checked by reading, not compiled, not run on hardware
//
#include <hip/hip_runtime.h>

typedef __attribute__((ext_vector_type(16))) __bf16 v16bf;
typedef __attribute__((ext_vector_type(8)))  float  v8f;

union Frag {
  v16bf v;
  unsigned int u[8];
};

__device__ __forceinline__ unsigned short f2bf(float f) {
  unsigned int u = __float_as_uint(f);
  u += 0x7fffu + ((u >> 16) & 1u);        // round-to-nearest-even
  return (unsigned short)(u >> 16);
}

__device__ __forceinline__ v8f wmma_bf16(v16bf a, v16bf b, v8f c) {
  // D = A(16x32 bf16) * B(32x16 bf16) + C(16x16 f32)
  return __builtin_amdgcn_wmma_f32_16x16x32_bf16(false, a, false, b, (short)0, c,
                                                 false, false);
}

// ---- 16-lane row reductions via DPP16 (no LDS, no ds_bpermute) ----
// Butterfly: quad_perm xor1 (0xB1), quad_perm xor2 (0x4E),
//            row_half_mirror (0x141, crosses 4-groups), row_mirror (0x140, crosses 8-groups)
template <int CTRL>
__device__ __forceinline__ float dpp_mv(float x) {
  return __int_as_float(__builtin_amdgcn_update_dpp(
      0, __float_as_int(x), CTRL, 0xF, 0xF, true));
}
__device__ __forceinline__ float row_max16(float x) {
  x = fmaxf(x, dpp_mv<0xB1>(x));
  x = fmaxf(x, dpp_mv<0x4E>(x));
  x = fmaxf(x, dpp_mv<0x141>(x));
  x = fmaxf(x, dpp_mv<0x140>(x));
  return x;
}
__device__ __forceinline__ float row_sum16(float x) {
  x += dpp_mv<0xB1>(x);
  x += dpp_mv<0x4E>(x);
  x += dpp_mv<0x141>(x);
  x += dpp_mv<0x140>(x);
  return x;
}

// ---------------- f32 -> bf16 convert ----------------
__global__ void cvt_bf16_kernel(const float* __restrict__ src,
                                unsigned short* __restrict__ dst, int n) {
  int i = blockIdx.x * blockDim.x + threadIdx.x;
  if (i < n) dst[i] = f2bf(src[i]);
}

// ---------------- LayerNorm over C=768, output bf16 ----------------
__global__ void __launch_bounds__(256) ln_kernel(
    const float* __restrict__ x, const float* __restrict__ g,
    const float* __restrict__ b, unsigned short* __restrict__ out) {
  const int C = 768;
  int row = blockIdx.x, t = threadIdx.x;
  const float* xr = x + (size_t)row * C;
  float loc[3];
  float s = 0.f, s2 = 0.f;
#pragma unroll
  for (int i = 0; i < 3; ++i) {
    float v = xr[t + i * 256];
    loc[i] = v; s += v; s2 += v * v;
  }
  __shared__ float r1[256], r2[256];
  r1[t] = s; r2[t] = s2;
  __syncthreads();
  for (int o = 128; o > 0; o >>= 1) {
    if (t < o) { r1[t] += r1[t + o]; r2[t] += r2[t + o]; }
    __syncthreads();
  }
  float mu = r1[0] * (1.f / 768.f);
  float var = r2[0] * (1.f / 768.f) - mu * mu;
  float rstd = rsqrtf(var + 1e-5f);
  unsigned short* orow = out + (size_t)row * C;
#pragma unroll
  for (int i = 0; i < 3; ++i) {
    int c = t + i * 256;
    orow[c] = f2bf((loc[i] - mu) * rstd * g[c] + b[c]);
  }
}

// ---------------- bf16 WMMA GEMM: C[M,N] = A[M,K] @ W[K,N] + bias ----------------
// EPI: 0 = bias -> bf16 out | 1 = bias + residual(f32) -> f32 out | 2 = bias + gelu -> bf16 out
// Block: 256 threads (8 waves), tile 128x64, per-wave 32x32 (2x2 WMMA tiles), BK=32.
template <int EPI>
__global__ void __launch_bounds__(256, 1) gemm_bf16_kernel(
    const unsigned short* __restrict__ A,
    const unsigned short* __restrict__ W,
    const float* __restrict__ bias,
    const float* __restrict__ residual,
    unsigned short* __restrict__ outB,
    float* __restrict__ outF,
    int M, int N, int K) {
  __shared__ __align__(16) unsigned short As[128 * 32];   // [row][k]
  __shared__ __align__(16) unsigned short Bs[64 * 32];    // [n][k] (transposed)
  const int tid  = threadIdx.x;
  const int lane = tid & 31, wid = tid >> 5;
  const int half = lane >> 4, ln = lane & 15;
  const int rowBase = blockIdx.y * 128;
  const int colBase = blockIdx.x * 64;
  const int wm = (wid >> 1) * 32;
  const int wn = (wid & 1) * 32;

  v8f acc[2][2];
#pragma unroll
  for (int i = 0; i < 2; ++i)
#pragma unroll
    for (int j = 0; j < 2; ++j) acc[i][j] = (v8f){};

  for (int k0 = 0; k0 < K; k0 += 32) {
    __syncthreads();
    // A tile 128x32 bf16: 512 16-byte chunks across 256 threads
    for (int i = tid; i < 512; i += 256) {
      int r = i >> 2, cc = (i & 3) * 8;
      *(uint4*)&As[r * 32 + cc] =
          *(const uint4*)&A[(size_t)(rowBase + r) * K + k0 + cc];
    }
    // W tile 32x64 bf16, stored transposed Bs[n][k] so K-pairs are contiguous
    for (int i = tid; i < 1024; i += 256) {
      int k = i >> 5, n2 = (i & 31) * 2;
      unsigned int val =
          *(const unsigned int*)&W[(size_t)(k0 + k) * N + colBase + n2];
      Bs[n2 * 32 + k]       = (unsigned short)(val & 0xffffu);
      Bs[(n2 + 1) * 32 + k] = (unsigned short)(val >> 16);
    }
    __syncthreads();

    Frag a[2], b[2];
#pragma unroll
    for (int mt = 0; mt < 2; ++mt) {
      int arow = wm + mt * 16 + ln;
#pragma unroll
      for (int j = 0; j < 8; ++j) {
        int kk = half * 16 + 2 * j;   // contiguous k order: k = 16*half + s
        a[mt].u[j] = *(const unsigned int*)&As[arow * 32 + kk];
      }
    }
#pragma unroll
    for (int nt = 0; nt < 2; ++nt) {
      int bcol = wn + nt * 16 + ln;
#pragma unroll
      for (int j = 0; j < 8; ++j)                            // contiguous k order, same as A
        b[nt].u[j] = *(const unsigned int*)&Bs[bcol * 32 + half * 16 + 2 * j];
    }
#pragma unroll
    for (int mt = 0; mt < 2; ++mt)
#pragma unroll
      for (int nt = 0; nt < 2; ++nt)
        acc[mt][nt] = wmma_bf16(a[mt].v, b[nt].v, acc[mt][nt]);
  }

  // Epilogue (C layout: vgpr r -> rows r / r+8 by lane half, col = lane&15)
#pragma unroll
  for (int mt = 0; mt < 2; ++mt)
#pragma unroll
    for (int nt = 0; nt < 2; ++nt)
#pragma unroll
      for (int r = 0; r < 8; ++r) {
        int row = rowBase + wm + mt * 16 + r + 8 * half;
        int col = colBase + wn + nt * 16 + ln;
        float v = acc[mt][nt][r] + bias[col];
        if (EPI == 2) {
          float c3 = v * v * v;
          v = 0.5f * v * (1.f + tanhf(0.7978845608028654f * (v + 0.044715f * c3)));
        }
        size_t idx = (size_t)row * N + col;
        if (EPI == 1) outF[idx] = residual[idx] + v;
        else          outB[idx] = f2bf(v);
      }
}

// ---------------- causal flash attention, one wave per 16-row Q tile ----------------
// qkv: bf16 [B*T, 2304] (q|k|v, head h at col h*64); y: bf16 [B*T, 768]
__global__ void __launch_bounds__(128, 1) attn_kernel(
    const unsigned short* __restrict__ qkv, unsigned short* __restrict__ y) {
  const int T = 1024, CQ = 2304, CY = 768;
  const int lane = threadIdx.x & 31, wid = threadIdx.x >> 5;
  const int half = lane >> 4, ln = lane & 15;
  const int head = blockIdx.y, batch = blockIdx.z;
  const int q0 = (blockIdx.x * 4 + wid) * 16;
  const size_t base = (size_t)batch * T * CQ;
  const unsigned short* Q = qkv + base + head * 64;
  const unsigned short* K = qkv + base + 768 + head * 64;
  const unsigned short* V = qkv + base + 1536 + head * 64;

  __shared__ __align__(8) unsigned short Plds[4 * 16 * 32];  // per-wave P staging
  unsigned short* myP = Plds + wid * 512;

  // Q fragments for D-chunks 0 and 32 (A layout, D-pairs contiguous in memory)
  Frag aq[2];
#pragma unroll
  for (int dc = 0; dc < 2; ++dc)
#pragma unroll
    for (int j = 0; j < 8; ++j) {
      int d = dc * 32 + half * 16 + 2 * j;
      aq[dc].u[j] = *(const unsigned int*)&Q[(size_t)(q0 + ln) * CQ + d];
    }

  v8f acc[4];
#pragma unroll
  for (int i = 0; i < 4; ++i) acc[i] = (v8f){};
  float m_i[8], l_i[8];
#pragma unroll
  for (int r = 0; r < 8; ++r) { m_i[r] = -1e30f; l_i[r] = 0.f; }

  for (int kb = 0; kb < q0 + 16; kb += 32) {         // causal: keys <= q0+15
    // ---- scores S = Q @ K^T * 1/sqrt(64), two 16x16 tiles over 32 keys ----
    v8f s[2];
#pragma unroll
    for (int st = 0; st < 2; ++st) {
      v8f sc = (v8f){};
#pragma unroll
      for (int dc = 0; dc < 2; ++dc) {
        Frag bk;                                     // B-frag: n=key, K=d
        int key = kb + st * 16 + ln;
#pragma unroll
        for (int j = 0; j < 8; ++j) {
          int d = dc * 32 + half * 16 + 2 * j;
          bk.u[j] = *(const unsigned int*)&K[(size_t)key * CQ + d];
        }
        sc = wmma_bf16(aq[dc].v, bk.v, sc);
      }
      s[st] = sc;
    }
    // ---- mask + online softmax (row r+8*half lives in this lane's 16-group) ----
#pragma unroll
    for (int r = 0; r < 8; ++r) {
      int rowg = q0 + r + 8 * half;
      float s0 = s[0][r] * 0.125f; if (kb + ln > rowg)      s0 = -1e30f;
      float s1 = s[1][r] * 0.125f; if (kb + 16 + ln > rowg) s1 = -1e30f;
      float mx = row_max16(fmaxf(s0, s1));           // DPP butterfly, no LDS
      float mn = fmaxf(m_i[r], mx);
      float alpha = __expf(m_i[r] - mn);
      float p0 = __expf(s0 - mn), p1 = __expf(s1 - mn);
      float ps = row_sum16(p0 + p1);                 // DPP butterfly, no LDS
      m_i[r] = mn;
      l_i[r] = l_i[r] * alpha + ps;
#pragma unroll
      for (int dt = 0; dt < 4; ++dt) acc[dt][r] *= alpha;
      myP[(r + 8 * half) * 32 + ln]      = f2bf(p0);   // C-layout -> LDS tile
      myP[(r + 8 * half) * 32 + 16 + ln] = f2bf(p1);
    }
    __asm__ volatile("" ::: "memory");
    __builtin_amdgcn_wave_barrier();                 // cross-lane LDS transpose, same wave
    Frag pa;                                          // P in A layout
#pragma unroll
    for (int j = 0; j < 8; ++j) {
      int kk = half * 16 + 2 * j;
      pa.u[j] = *(const unsigned int*)&myP[ln * 32 + kk];
    }
    __builtin_amdgcn_wave_barrier();
    __asm__ volatile("" ::: "memory");
    // ---- O += P @ V (4 d-tiles of 16) ----
#pragma unroll
    for (int dt = 0; dt < 4; ++dt) {
      Frag bv;                                        // B-frag: n=d, K=key
#pragma unroll
      for (int j = 0; j < 8; ++j) {
        int key = kb + half * 16 + 2 * j;
        unsigned int lo = V[(size_t)key * CQ + dt * 16 + ln];
        unsigned int hi = V[(size_t)(key + 1) * CQ + dt * 16 + ln];
        bv.u[j] = lo | (hi << 16);
      }
      acc[dt] = wmma_bf16(pa.v, bv.v, acc[dt]);
    }
  }
  // ---- normalize and store y (bf16) ----
#pragma unroll
  for (int dt = 0; dt < 4; ++dt)
#pragma unroll
    for (int r = 0; r < 8; ++r) {
      int rowg = q0 + r + 8 * half;
      float v = acc[dt][r] / l_i[r];
      y[((size_t)batch * T + rowg) * CY + head * 64 + dt * 16 + ln] = f2bf(v);
    }
}

extern "C" void kernel_launch(void* const* d_in, const int* in_sizes, int n_in,
                              void* d_out, int out_size, void* d_ws, size_t ws_size,
                              hipStream_t stream) {
  (void)in_sizes; (void)n_in; (void)out_size; (void)ws_size;
  const float* x      = (const float*)d_in[0];
  const float* ln1_g  = (const float*)d_in[1];
  const float* ln1_b  = (const float*)d_in[2];
  const float* w_attn = (const float*)d_in[3];
  const float* b_attn = (const float*)d_in[4];
  const float* w_proj = (const float*)d_in[5];
  const float* b_proj = (const float*)d_in[6];
  const float* ln2_g  = (const float*)d_in[7];
  const float* ln2_b  = (const float*)d_in[8];
  const float* w_fc1  = (const float*)d_in[9];
  const float* b_fc1  = (const float*)d_in[10];
  const float* w_fc2  = (const float*)d_in[11];
  const float* b_fc2  = (const float*)d_in[12];
  float* out = (float*)d_out;

  const int BT = 8 * 1024;  // 8192 token rows
  char* ws = (char*)d_ws;
  size_t off = 0;
  auto take = [&](size_t bytes) -> char* {
    char* p = ws + off;
    off += (bytes + 255) & ~(size_t)255;
    return p;
  };
  unsigned short* wA  = (unsigned short*)take((size_t)768 * 2304 * 2);
  unsigned short* wP  = (unsigned short*)take((size_t)768 * 768 * 2);
  unsigned short* w1  = (unsigned short*)take((size_t)768 * 3072 * 2);
  unsigned short* w2  = (unsigned short*)take((size_t)3072 * 768 * 2);
  unsigned short* h1  = (unsigned short*)take((size_t)BT * 768 * 2);   // LN out (reused for LN2)
  unsigned short* qkv = (unsigned short*)take((size_t)BT * 2304 * 2);
  unsigned short* yb  = (unsigned short*)take((size_t)BT * 768 * 2);
  // fc1 output (8192x3072 bf16 = 50.3MB) aliases qkv+yb, both dead by then.
  unsigned short* hmid = qkv;

  auto cvt = [&](const float* s, unsigned short* d, int n) {
    cvt_bf16_kernel<<<(n + 255) / 256, 256, 0, stream>>>(s, d, n);
  };
  cvt(w_attn, wA, 768 * 2304);
  cvt(w_proj, wP, 768 * 768);
  cvt(w_fc1,  w1, 768 * 3072);
  cvt(w_fc2,  w2, 3072 * 768);

  // h1 = bf16(LN1(x))
  ln_kernel<<<BT, 256, 0, stream>>>(x, ln1_g, ln1_b, h1);
  // qkv = bf16(h1 @ w_attn + b_attn)
  gemm_bf16_kernel<0><<<dim3(2304 / 64, BT / 128), 256, 0, stream>>>(
      h1, wA, b_attn, nullptr, qkv, nullptr, BT, 2304, 768);
  // yb = attention(qkv)
  attn_kernel<<<dim3(16, 12, 8), 128, 0, stream>>>(qkv, yb);
  // out = x + yb @ w_proj + b_proj   (f32)
  gemm_bf16_kernel<1><<<dim3(768 / 64, BT / 128), 256, 0, stream>>>(
      yb, wP, b_proj, x, nullptr, out, BT, 768, 768);
  // h1 = bf16(LN2(out))
  ln_kernel<<<BT, 256, 0, stream>>>(out, ln2_g, ln2_b, h1);
  // hmid = bf16(gelu(h1 @ w_fc1 + b_fc1))
  gemm_bf16_kernel<2><<<dim3(3072 / 64, BT / 128), 256, 0, stream>>>(
      h1, w1, b_fc1, nullptr, hmid, nullptr, BT, 3072, 768);
  // out = out + hmid @ w_fc2 + b_fc2  (f32)
  gemm_bf16_kernel<1><<<dim3(768 / 64, BT / 128), 256, 0, stream>>>(
      hmid, w2, b_fc2, out, nullptr, out, BT, 768, 3072);
}
